// RecursiveCognitiveBlock_77232101916933
// MI455X (gfx1250) — matrix operand order repaired and checked by reading, not compiled, not run on hardware
//
#include <hip/hip_runtime.h>
#include <stdint.h>

typedef __attribute__((ext_vector_type(16))) __bf16       v16bf;
typedef __attribute__((ext_vector_type(8)))  float        v8f;
typedef __attribute__((ext_vector_type(8)))  unsigned int v8u;

#define D_DIM 1024
#define F_DIM 2048
#define E_NUM 8
#define MT 128   // M-tile rows per workgroup (8 waves x 16)
#define NT 64    // N-tile cols per workgroup (4 x 16)

#if __has_builtin(__builtin_amdgcn_global_load_async_to_lds_b128)
#define HAVE_ASYNC_LDS 1
typedef int v4ib __attribute__((vector_size(16)));
typedef __attribute__((address_space(1))) v4ib* as1p;   // prints as 'int4 __device__ *'
typedef __attribute__((address_space(3))) v4ib* as3p;   // LDS-side payload pointer
#else
#define HAVE_ASYNC_LDS 0
#endif

__device__ __forceinline__ void wait_async0() {
#if __has_builtin(__builtin_amdgcn_s_wait_asynccnt)
  __builtin_amdgcn_s_wait_asynccnt(0);
#else
  asm volatile("s_wait_asynccnt 0x0" ::: "memory");
#endif
}

__device__ __forceinline__ unsigned short f2bf(float x) {
  unsigned int u = __builtin_bit_cast(unsigned int, x);
  u += 0x7FFFu + ((u >> 16) & 1u);   // round-to-nearest-even
  return (unsigned short)(u >> 16);
}

// Copy 16B global -> LDS, async DMA when the toolchain exposes it.
__device__ __forceinline__ void copy16_to_lds(const unsigned short* gsrc,
                                              unsigned int* ldst) {
#if HAVE_ASYNC_LDS
  __builtin_amdgcn_global_load_async_to_lds_b128(
      (as1p)(unsigned long long)gsrc,   // flat == global address, value-preserving
      (as3p)(void*)ldst,                // C-style cast -> addrspacecast to LDS
      0, 0);
#else
  *(uint4*)ldst = *(const uint4*)gsrc;
#endif
}

// ---------------------------------------------------------------------------
// Kernel 0: reset slot arrays, expert counts, and the zero padding row
// ---------------------------------------------------------------------------
__global__ void moe_init(int* __restrict__ tokArr, float* __restrict__ cwArr,
                         int* __restrict__ counts, unsigned short* __restrict__ zpad,
                         int cap) {
  int i = blockIdx.x * blockDim.x + threadIdx.x;
  if (i < cap) { tokArr[i] = -1; cwArr[i] = 0.0f; }
  if (i < E_NUM) counts[i] = 0;
  if (i < D_DIM) zpad[i] = 0;
}

// ---------------------------------------------------------------------------
// Kernel 1: RMSNorm + bf16 z_norm + router softmax/top2 + residual base copy
// One 256-thread block per token (4 floats / thread over D=1024).
// ---------------------------------------------------------------------------
__global__ __launch_bounds__(256) void moe_norm_router(
    const float* __restrict__ zin, const float* __restrict__ nw,
    const float* __restrict__ rw, unsigned short* __restrict__ zn,
    float* __restrict__ accout, int* __restrict__ counts,
    int* __restrict__ tkidx, float* __restrict__ tkw) {
  const int t = blockIdx.x;
  const int tid = threadIdx.x;
  const float* zr = zin + (size_t)t * D_DIM;
  float4 v = *(const float4*)(zr + tid * 4);

  __shared__ float red[256];
  __shared__ float rp[256 * E_NUM];

  red[tid] = v.x * v.x + v.y * v.y + v.z * v.z + v.w * v.w;
  __syncthreads();
  for (int o = 128; o > 0; o >>= 1) {
    if (tid < o) red[tid] += red[tid + o];
    __syncthreads();
  }
  const float rms = rsqrtf(red[0] * (1.0f / (float)D_DIM) + 1e-6f);

  float4 w = *(const float4*)(nw + tid * 4);
  float zn4[4] = { v.x * rms * w.x, v.y * rms * w.y,
                   v.z * rms * w.z, v.w * rms * w.w };

  // bf16 z_norm (GEMM A operand)
  uint2 zp;
  zp.x = (unsigned int)f2bf(zn4[0]) | ((unsigned int)f2bf(zn4[1]) << 16);
  zp.y = (unsigned int)f2bf(zn4[2]) | ((unsigned int)f2bf(zn4[3]) << 16);
  *(uint2*)(zn + (size_t)t * D_DIM + tid * 4) = zp;

  // residual base (out = z + mixed): initialize accumulator with z
  *(float4*)(accout + (size_t)t * D_DIM + tid * 4) = v;

  // router logits: partial dot per expert
  float lp[E_NUM];
#pragma unroll
  for (int e = 0; e < E_NUM; ++e) lp[e] = 0.0f;
#pragma unroll
  for (int j = 0; j < 4; ++j) {
    const float zv = zn4[j];
    const float* rr = rw + (size_t)(tid * 4 + j) * E_NUM;
#pragma unroll
    for (int e = 0; e < E_NUM; ++e) lp[e] += zv * rr[e];
  }
#pragma unroll
  for (int e = 0; e < E_NUM; ++e) rp[tid * E_NUM + e] = lp[e];
  __syncthreads();
  for (int o = 128; o > 0; o >>= 1) {
    if (tid < o) {
#pragma unroll
      for (int e = 0; e < E_NUM; ++e)
        rp[tid * E_NUM + e] += rp[(tid + o) * E_NUM + e];
    }
    __syncthreads();
  }

  if (tid == 0) {
    float l[E_NUM], p[E_NUM];
    float m = -1e30f;
#pragma unroll
    for (int e = 0; e < E_NUM; ++e) { l[e] = rp[e]; m = fmaxf(m, l[e]); }
    float s = 0.0f;
#pragma unroll
    for (int e = 0; e < E_NUM; ++e) { p[e] = __expf(l[e] - m); s += p[e]; }
    const float inv = 1.0f / s;
#pragma unroll
    for (int e = 0; e < E_NUM; ++e) p[e] *= inv;
    int i1 = 0;
    for (int e = 1; e < E_NUM; ++e) if (p[e] > p[i1]) i1 = e;
    int i2 = (i1 == 0) ? 1 : 0;
    for (int e = 0; e < E_NUM; ++e) if (e != i1 && p[e] > p[i2]) i2 = e;
    float a = p[i1], b = p[i2];
    const float sw = 1.0f / (a + b + 1e-9f);
    a *= sw; b *= sw;
    tkidx[t * 2 + 0] = i1; tkidx[t * 2 + 1] = i2;
    tkw[t * 2 + 0] = a;    tkw[t * 2 + 1] = b;
    atomicAdd(&counts[i1], 1);
    atomicAdd(&counts[i2], 1);
  }
}

// ---------------------------------------------------------------------------
// Kernel 2: tiny scan -> padded per-expert offsets + tile schedule table
// ---------------------------------------------------------------------------
__global__ void moe_scan(const int* __restrict__ counts, int* __restrict__ cursors,
                         int* __restrict__ tileExp, int* __restrict__ tileBase,
                         int maxtiles) {
  if (threadIdx.x != 0 || blockIdx.x != 0) return;
  int cum = 0, ti = 0;
  for (int e = 0; e < E_NUM; ++e) {
    cursors[e] = cum;
    const int ntile = (counts[e] + MT - 1) / MT;
    for (int i = 0; i < ntile; ++i) { tileExp[ti] = e; tileBase[ti] = cum + i * MT; ++ti; }
    cum += ntile * MT;
  }
  for (; ti < maxtiles; ++ti) { tileExp[ti] = -1; tileBase[ti] = 0; }
}

// ---------------------------------------------------------------------------
// Kernel 3: compact (token, weight) slots per expert
// ---------------------------------------------------------------------------
__global__ void moe_fill(const int* __restrict__ tkidx, const float* __restrict__ tkw,
                         int* __restrict__ cursors, int* __restrict__ tokArr,
                         float* __restrict__ cwArr, int T) {
  int t = blockIdx.x * blockDim.x + threadIdx.x;
  if (t >= T) return;
#pragma unroll
  for (int k = 0; k < 2; ++k) {
    const int e = tkidx[t * 2 + k];
    const int slot = atomicAdd(&cursors[e], 1);
    tokArr[slot] = t;
    cwArr[slot] = tkw[t * 2 + k];
  }
}

// ---------------------------------------------------------------------------
// WMMA GEMM stage 1: H = silu(X @ W1) * (X @ W3), gathered rows, bf16 out
// Grid: (F/NT, maxtiles). 8 waves; wave w owns rows [w*16, w*16+16), all NT cols.
// A tile rides the async global->LDS path; B tiles go through VGPRs for the
// fp32->bf16 conversion and overlap the A-tile DMA.
// ---------------------------------------------------------------------------
__global__ __launch_bounds__(256) void moe_mlp1(
    const unsigned short* __restrict__ zn,
    const float* __restrict__ w1g, const float* __restrict__ w3g,
    const int* __restrict__ tokArr, const unsigned short* __restrict__ zpad,
    const int* __restrict__ tileExp, const int* __restrict__ tileBase,
    unsigned short* __restrict__ Hbuf) {
  const int tile = blockIdx.y;
  const int e = tileExp[tile];
  if (e < 0) return;
  const int rbase = tileBase[tile];
  const int n0 = blockIdx.x * NT;
  const float* __restrict__ W1 = w1g + (size_t)e * D_DIM * F_DIM;
  const float* __restrict__ W3 = w3g + (size_t)e * D_DIM * F_DIM;

  __shared__ __align__(16) unsigned int sA[MT * 16];   // 128 rows x 32 bf16 (pair-packed)
  __shared__ __align__(16) unsigned int sB1[NT * 16];  // [n][kpair] transposed pair-packed
  __shared__ __align__(16) unsigned int sB3[NT * 16];

  const int tid = threadIdx.x;
  const int wave = tid >> 5;
  const int lane = tid & 31;
  const int hi = lane >> 4;
  const int lm = lane & 15;

  v8f acc1[4], acc3[4];
#pragma unroll
  for (int i = 0; i < 4; ++i)
#pragma unroll
    for (int j = 0; j < 8; ++j) { acc1[i][j] = 0.0f; acc3[i][j] = 0.0f; }

  const int arow = tid >> 1;
  const int ahalf = tid & 1;
  const int tok = tokArr[rbase + arow];
  // padding rows read a zeroed 2KB page -> branch-free, EXEC stays all-1s
  const unsigned short* asrc =
      ((tok >= 0) ? (zn + (size_t)tok * D_DIM) : zpad) + ahalf * 16;
  unsigned int* adst = &sA[arow * 16 + ahalf * 8];

  for (int k0 = 0; k0 < D_DIM; k0 += 32) {
    __syncthreads();
    // ---- A tile: 2 x 16B per thread, async DMA into LDS
    copy16_to_lds(asrc + k0, adst);
    copy16_to_lds(asrc + k0 + 8, adst + 4);

    // ---- B tiles: fp32 -> bf16 convert, store transposed pair-packed
#pragma unroll
    for (int m = 0; m < 2; ++m) {
      const int q = tid * 2 + m;
      const int k = q >> 4, g = q & 15;
      const size_t goff = (size_t)(k0 + k) * F_DIM + n0 + g * 4;
      const float4 b1 = *(const float4*)(W1 + goff);
      const float4 b3 = *(const float4*)(W3 + goff);
      unsigned short* t1 = (unsigned short*)sB1;
      unsigned short* t3 = (unsigned short*)sB3;
      const float b1a[4] = {b1.x, b1.y, b1.z, b1.w};
      const float b3a[4] = {b3.x, b3.y, b3.z, b3.w};
#pragma unroll
      for (int j = 0; j < 4; ++j) {
        const int n = g * 4 + j;
        t1[n * 32 + k] = f2bf(b1a[j]);
        t3[n * 32 + k] = f2bf(b3a[j]);
      }
    }
    if (k0 + 32 < D_DIM) {  // warm L2 for the next K strip
      __builtin_prefetch(W1 + (size_t)(k0 + 32) * F_DIM + n0, 0, 1);
      __builtin_prefetch(W3 + (size_t)(k0 + 32) * F_DIM + n0, 0, 1);
    }
#if HAVE_ASYNC_LDS
    wait_async0();
#endif
    __syncthreads();

    // ---- A fragment (contiguous k = 16*hi + s, matching B)
    v8u av;
    const int rb = (wave * 16 + lm) * 16 + hi * 8;
#pragma unroll
    for (int v = 0; v < 8; ++v) av[v] = sA[rb + v];
    const v16bf Afrag = __builtin_bit_cast(v16bf, av);

#pragma unroll
    for (int nt = 0; nt < 4; ++nt) {
      const int nb = (nt * 16 + lm) * 16 + hi * 8;
      v8u bv;
#pragma unroll
      for (int v = 0; v < 8; ++v) bv[v] = sB1[nb + v];
      acc1[nt] = __builtin_amdgcn_wmma_f32_16x16x32_bf16(
          false, Afrag, false, __builtin_bit_cast(v16bf, bv),
          (short)0, acc1[nt], false, false);
#pragma unroll
      for (int v = 0; v < 8; ++v) bv[v] = sB3[nb + v];
      acc3[nt] = __builtin_amdgcn_wmma_f32_16x16x32_bf16(
          false, Afrag, false, __builtin_bit_cast(v16bf, bv),
          (short)0, acc3[nt], false, false);
    }
  }

  // ---- fused SwiGLU epilogue, bf16 store of H
#pragma unroll
  for (int nt = 0; nt < 4; ++nt) {
#pragma unroll
    for (int v = 0; v < 8; ++v) {
      const float a = acc1[nt][v];
      const float hval = (a / (1.0f + __expf(-a))) * acc3[nt][v];
      const int row = wave * 16 + v + hi * 8;
      const int col = n0 + nt * 16 + lm;
      Hbuf[(size_t)(rbase + row) * F_DIM + col] = f2bf(hval);
    }
  }
}

// ---------------------------------------------------------------------------
// WMMA GEMM stage 2: Y = H @ W2, scatter-add cw * Y into accumulator
// Grid: (D/NT, maxtiles).
// ---------------------------------------------------------------------------
__global__ __launch_bounds__(256) void moe_mlp2(
    const unsigned short* __restrict__ Hbuf,
    const float* __restrict__ w2g,
    const int* __restrict__ tokArr, const float* __restrict__ cwArr,
    const int* __restrict__ tileExp, const int* __restrict__ tileBase,
    float* __restrict__ accout) {
  const int tile = blockIdx.y;
  const int e = tileExp[tile];
  if (e < 0) return;
  const int rbase = tileBase[tile];
  const int n0 = blockIdx.x * NT;
  const float* __restrict__ W2 = w2g + (size_t)e * F_DIM * D_DIM;

  __shared__ __align__(16) unsigned int sA[MT * 16];
  __shared__ __align__(16) unsigned int sB[NT * 16];

  const int tid = threadIdx.x;
  const int wave = tid >> 5;
  const int lane = tid & 31;
  const int hi = lane >> 4;
  const int lm = lane & 15;

  v8f acc[4];
#pragma unroll
  for (int i = 0; i < 4; ++i)
#pragma unroll
    for (int j = 0; j < 8; ++j) acc[i][j] = 0.0f;

  const int arow = tid >> 1;
  const int ahalf = tid & 1;
  const unsigned short* asrc = Hbuf + (size_t)(rbase + arow) * F_DIM + ahalf * 16;
  unsigned int* adst = &sA[arow * 16 + ahalf * 8];

  for (int k0 = 0; k0 < F_DIM; k0 += 32) {
    __syncthreads();
    // ---- A tile: async DMA of H rows into LDS
    copy16_to_lds(asrc + k0, adst);
    copy16_to_lds(asrc + k0 + 8, adst + 4);

#pragma unroll
    for (int m = 0; m < 2; ++m) {
      const int q = tid * 2 + m;
      const int k = q >> 4, g = q & 15;
      const float4 b = *(const float4*)(W2 + (size_t)(k0 + k) * D_DIM + n0 + g * 4);
      unsigned short* tb = (unsigned short*)sB;
      const float ba[4] = {b.x, b.y, b.z, b.w};
#pragma unroll
      for (int j = 0; j < 4; ++j) tb[(g * 4 + j) * 32 + k] = f2bf(ba[j]);
    }
    if (k0 + 32 < F_DIM)
      __builtin_prefetch(W2 + (size_t)(k0 + 32) * D_DIM + n0, 0, 1);
#if HAVE_ASYNC_LDS
    wait_async0();
#endif
    __syncthreads();

    v8u av;
    const int rb = (wave * 16 + lm) * 16 + hi * 8;
#pragma unroll
    for (int v = 0; v < 8; ++v) av[v] = sA[rb + v];
    const v16bf Afrag = __builtin_bit_cast(v16bf, av);

#pragma unroll
    for (int nt = 0; nt < 4; ++nt) {
      const int nb = (nt * 16 + lm) * 16 + hi * 8;
      v8u bv;
#pragma unroll
      for (int v = 0; v < 8; ++v) bv[v] = sB[nb + v];
      acc[nt] = __builtin_amdgcn_wmma_f32_16x16x32_bf16(
          false, Afrag, false, __builtin_bit_cast(v16bf, bv),
          (short)0, acc[nt], false, false);
    }
  }

  // ---- scatter: out[tok] += cw * y   (exactly 2 adds per element -> order-safe)
  int toks[8];
  float cws[8];
#pragma unroll
  for (int v = 0; v < 8; ++v) {
    const int row = rbase + wave * 16 + v + hi * 8;
    toks[v] = tokArr[row];
    cws[v] = cwArr[row];
  }
#pragma unroll
  for (int nt = 0; nt < 4; ++nt) {
    const int col = n0 + nt * 16 + lm;
#pragma unroll
    for (int v = 0; v < 8; ++v) {
      if (toks[v] >= 0)
        unsafeAtomicAdd(&accout[(size_t)toks[v] * D_DIM + col], cws[v] * acc[nt][v]);
    }
  }
}

// ---------------------------------------------------------------------------
extern "C" void kernel_launch(void* const* d_in, const int* in_sizes, int n_in,
                              void* d_out, int out_size, void* d_ws, size_t ws_size,
                              hipStream_t stream) {
  const float* z      = (const float*)d_in[0];
  const float* norm_w = (const float*)d_in[1];
  const float* rw     = (const float*)d_in[2];
  const float* w1     = (const float*)d_in[3];
  const float* w3     = (const float*)d_in[4];
  const float* w2     = (const float*)d_in[5];
  float* out = (float*)d_out;

  const int T = in_sizes[0] / D_DIM;               // 8192 tokens
  const int CAP = 2 * T + E_NUM * MT;              // padded slot capacity
  const int MAXTILES = (2 * T) / MT + E_NUM;       // worst-case M tiles

  char* ws = (char*)d_ws;
  size_t off = 0;
  auto carve = [&](size_t bytes) -> void* {
    void* p = ws + off;
    off += (bytes + 255) & ~(size_t)255;
    return p;
  };
  unsigned short* zn     = (unsigned short*)carve((size_t)T * D_DIM * 2);
  float*          accbuf = (float*)carve((size_t)T * D_DIM * 4);
  unsigned short* Hbuf   = (unsigned short*)carve((size_t)CAP * F_DIM * 2);
  int*   tokArr  = (int*)carve((size_t)CAP * 4);
  float* cwArr   = (float*)carve((size_t)CAP * 4);
  int*   tkidx   = (int*)carve((size_t)T * 2 * 4);
  float* tkw     = (float*)carve((size_t)T * 2 * 4);
  int*   counts  = (int*)carve(64);
  int*   cursors = (int*)carve(64);
  int*   tileExp = (int*)carve((size_t)MAXTILES * 4);
  int*   tileBase= (int*)carve((size_t)MAXTILES * 4);
  unsigned short* zpad = (unsigned short*)carve((size_t)D_DIM * 2);

  for (int it = 0; it < 2; ++it) {  // LOOPS = 2
    const float* zcur = (it == 0) ? z : accbuf;
    float* acc = (it == 0) ? accbuf : out;

    moe_init<<<dim3((CAP + 255) / 256), dim3(256), 0, stream>>>(tokArr, cwArr,
                                                                counts, zpad, CAP);
    moe_norm_router<<<dim3(T), dim3(256), 0, stream>>>(zcur, norm_w, rw, zn, acc,
                                                       counts, tkidx, tkw);
    moe_scan<<<dim3(1), dim3(1), 0, stream>>>(counts, cursors, tileExp, tileBase, MAXTILES);
    moe_fill<<<dim3(T / 256), dim3(256), 0, stream>>>(tkidx, tkw, cursors, tokArr, cwArr, T);
    moe_mlp1<<<dim3(F_DIM / NT, MAXTILES), dim3(256), 0, stream>>>(
        zn, w1, w3, tokArr, zpad, tileExp, tileBase, Hbuf);
    moe_mlp2<<<dim3(D_DIM / NT, MAXTILES), dim3(256), 0, stream>>>(
        Hbuf, w2, tokArr, cwArr, tileExp, tileBase, acc);
  }
}
